// SREXmodel_74586402063164
// MI455X (gfx1250) — compile-verified, matrix operand order checked by reading, not run on hardware
//
#include <hip/hip_runtime.h>
#include <hip/hip_bf16.h>

typedef _Float16 f16;
typedef __attribute__((ext_vector_type(16))) _Float16 v16h;
typedef __attribute__((ext_vector_type(8)))  _Float16 v8h;
typedef __attribute__((ext_vector_type(8)))  float    v8f;

// Problem constants
#define BB 8
#define NN 500
#define RR 20
#define FF 16
#define HH 8
#define CC 64
#define DD 512          // H*C
#define EE 64000
#define NT 4000         // B*N
#define EDG 68000       // E + NT self loops
#define D2 256          // D/2
#define PROWS 3040      // B*(R-1)*R
#define NOUT 60800      // B*(R-1)*R*R

// Workspace byte offsets (all 16B aligned)
#define OFF_H      0ull                   // 2*4000*512 f32   = 16,384,000
#define OFF_ASRC   16384000ull            // 2*4000*8  f32    = 256,000
#define OFF_ADST   16640000ull            // 256,000
#define OFF_MKEY   16896000ull            // 2*4000*8  u32    = 256,000
#define OFF_Z      17152000ull            // 256,000
#define OFF_EBUF   17408000ull            // 2*68000*8 f32    = 4,352,000
#define OFF_GOUT   21760000ull            // 2*4000*512 f32   = 16,384,000
#define OFF_R      38144000ull            // 2*8*20*512 f32   = 655,360
#define OFF_S      38799360ull            // 2*8*19*20*512 f16 = 6,225,920
#define OFF_P      51251200ull            // 2*3040*512 f32   = 12,451,840
#define OFF_W1H    63703040ull            // 1024*512 f16 (transposed [g][n][k]) = 1,048,576
#define OFF_W2H    64751616ull            // 512*256 f16 (transposed [n][k])     = 262,144

__device__ __forceinline__ float lrelu01(float v){ return v > 0.f ? v : 0.01f*v; }
__device__ __forceinline__ float lrelu02(float v){ return v > 0.f ? v : 0.2f*v; }

#define CAT16(lo, hi) __builtin_shufflevector(lo, hi, 0,1,2,3,4,5,6,7,8,9,10,11,12,13,14,15)

// load a 16-bit WMMA fragment: two contiguous 8-half runs at k0 and k0+16
__device__ __forceinline__ v16h frag16(const f16* base, int k0) {
    v8h lo = *(const v8h*)(base + k0);
    v8h hi = *(const v8h*)(base + k0 + 16);
    return CAT16(lo, hi);
}

// ---------------- K0: convert W1, W2 to transposed f16 ----------------
// W1hT[g][n][k] = W1[g*512+k][n];  W2hT[n][k] = W2[k][n]
__global__ void convert_weights(const float* __restrict__ W1, const float* __restrict__ W2,
                                f16* __restrict__ W1hT, f16* __restrict__ W2hT) {
    int i = blockIdx.x * blockDim.x + threadIdx.x;
    if (i < 1024*512) {
        int kp = i >> 9;          // 0..1023
        int n  = i & 511;
        int g  = kp >> 9;
        int k  = kp & 511;
        W1hT[((long)(g*DD + n))*DD + k] = (f16)W1[i];
    }
    if (i < 512*256) {
        int k = i >> 8;           // W2 row
        int n = i & 255;
        W2hT[(long)n*DD + k] = (f16)W2[i];
    }
}

// ---------------- K1: node linear + attention coefficients ----------------
__global__ void node_kernel(const float* __restrict__ x1, const float* __restrict__ x2,
                            const float* __restrict__ Wg, const float* __restrict__ att_s,
                            const float* __restrict__ att_d,
                            float* __restrict__ h, float* __restrict__ asrc, float* __restrict__ adst) {
    int blk = blockIdx.x;           // 0..7999
    int g = blk / NT;
    int n = blk % NT;
    const float* x = (g ? x2 : x1) + n * FF;
    __shared__ float xs[FF];
    __shared__ float as[HH], ad[HH];
    int t = threadIdx.x;
    if (t < FF) xs[t] = x[t];
    if (t < HH) { as[t] = 0.f; ad[t] = 0.f; }
    __syncthreads();
    float acc = 0.f;
    #pragma unroll
    for (int k = 0; k < FF; ++k) acc += xs[k] * Wg[k*DD + t];
    int node = g*NT + n;
    h[(size_t)node*DD + t] = acc;
    atomicAdd(&as[t >> 6], acc * att_s[t]);
    atomicAdd(&ad[t >> 6], acc * att_d[t]);
    __syncthreads();
    if (t < HH) {
        asrc[node*HH + t] = as[t];
        adst[node*HH + t] = ad[t];
    }
}

__device__ __forceinline__ void edge_pair(const int* ei, int e, int& src, int& dst) {
    if (e < EE) { src = ei[e]; dst = ei[EE + e]; }
    else        { src = dst = e - EE; }
}
__device__ __forceinline__ unsigned fkey(float f) {
    unsigned u = __float_as_uint(f);
    return (u & 0x80000000u) ? ~u : (u | 0x80000000u);
}
__device__ __forceinline__ float fdec(unsigned k) {
    return (k & 0x80000000u) ? __uint_as_float(k ^ 0x80000000u) : __uint_as_float(~k);
}

// ---------------- K2: edge scores + segment max ----------------
__global__ void edge_a_kernel(const int* __restrict__ ei1, const int* __restrict__ ei2,
                              const float* __restrict__ asrc, const float* __restrict__ adst,
                              float* __restrict__ ebuf, unsigned* __restrict__ mkey) {
    int idx = blockIdx.x * blockDim.x + threadIdx.x;
    if (idx >= 2*EDG*HH) return;
    int hd = idx & 7; int t = idx >> 3;
    int e = t % EDG;  int g = t / EDG;
    int src, dst; edge_pair(g ? ei2 : ei1, e, src, dst);
    float v = asrc[(g*NT + src)*HH + hd] + adst[(g*NT + dst)*HH + hd];
    v = lrelu02(v);
    ebuf[idx] = v;
    atomicMax(&mkey[(g*NT + dst)*HH + hd], fkey(v));
}

// ---------------- K3: exp + segment sum ----------------
__global__ void edge_b_kernel(const int* __restrict__ ei1, const int* __restrict__ ei2,
                              float* __restrict__ ebuf, const unsigned* __restrict__ mkey,
                              float* __restrict__ Z) {
    int idx = blockIdx.x * blockDim.x + threadIdx.x;
    if (idx >= 2*EDG*HH) return;
    int hd = idx & 7; int t = idx >> 3;
    int e = t % EDG;  int g = t / EDG;
    int src, dst; edge_pair(g ? ei2 : ei1, e, src, dst);
    float m = fdec(mkey[(g*NT + dst)*HH + hd]);
    float ex = __expf(ebuf[idx] - m);
    ebuf[idx] = ex;
    atomicAdd(&Z[(g*NT + dst)*HH + hd], ex);
}

// ---------------- K4: alpha * h[src] scatter ----------------
__global__ void edge_c_kernel(const int* __restrict__ ei1, const int* __restrict__ ei2,
                              const float* __restrict__ ebuf, const float* __restrict__ Z,
                              const float* __restrict__ h, float* __restrict__ gout) {
    long idx = (long)blockIdx.x * blockDim.x + threadIdx.x;
    if (idx >= (long)2*EDG*128) return;
    int q = (int)(idx & 127); long t = idx >> 7;
    int e = (int)(t % EDG);   int g = (int)(t / EDG);
    int src, dst; edge_pair(g ? ei2 : ei1, e, src, dst);
    int d = q * 4; int hd = d >> 6;
    float ex = ebuf[((long)(g*EDG + e))*HH + hd];
    float z  = Z[(g*NT + dst)*HH + hd];
    float alpha = ex / (z + 1e-16f);
    const float4 hv = *(const float4*)&h[((long)(g*NT + src))*DD + d];
    float* o = &gout[((long)(g*NT + dst))*DD + d];
    atomicAdd(o + 0, alpha * hv.x);
    atomicAdd(o + 1, alpha * hv.y);
    atomicAdd(o + 2, alpha * hv.z);
    atomicAdd(o + 3, alpha * hv.w);
}

// ---------------- K5: bias + lrelu + route pool ----------------
__global__ void emb_pool_kernel(const float* __restrict__ gout, const float* __restrict__ bg,
                                const int* __restrict__ rt1, const int* __restrict__ rt2,
                                float* __restrict__ r) {
    long idx = (long)blockIdx.x * blockDim.x + threadIdx.x;
    if (idx >= (long)2*NT*DD) return;
    int d = (int)(idx & (DD-1));
    int node = (int)(idx >> 9);
    int g = node / NT; int n = node % NT;
    float v = lrelu01(gout[idx] + bg[d]);
    int b = n / NN;
    int rt = (g ? rt2 : rt1)[n];
    atomicAdd(&r[(((long)(g*BB + b)*RR) + rt)*DD + d], v);
}

// ---------------- K6: cyclic window sums -> f16 [g][b][m][i][d] ----------------
__global__ void winsum_kernel(const float* __restrict__ r, f16* __restrict__ s) {
    int idx = blockIdx.x * blockDim.x + threadIdx.x;
    if (idx >= 2*BB*DD) return;
    int d = idx & (DD-1);
    int gb = idx >> 9;              // g*8 + b
    long base = (long)gb * RR * DD + d;
    for (int i = 0; i < RR; ++i) {
        float run = 0.f;
        for (int m = 0; m < RR-1; ++m) {
            run += r[base + (long)((i + m) % RR) * DD];
            s[((((long)gb*(RR-1) + m)*RR) + i)*DD + d] = (f16)run;
        }
    }
}

// ---------------- K7: P = S @ W1 (per-graph K offset) via WMMA ----------------
// S is f16 row-major; W1hT[g][n][k] column-major. One A fragment feeds two N-tiles.
__global__ __launch_bounds__(256) void pgemm_kernel(const f16* __restrict__ Sh,
                                                    const f16* __restrict__ W1hT,
                                                    float* __restrict__ P) {
    int wave = threadIdx.x >> 5;
    int lane = threadIdx.x & 31;
    int unit = blockIdx.x * 8 + wave;        // 2 * 190 * 16 = 6080 units
    int g   = unit / (190*16);
    int rem = unit % (190*16);
    int mt  = rem >> 4;
    int np  = rem & 15;                      // pair of N-tiles
    int half = lane >> 4;
    int lr   = lane & 15;
    int row0 = mt * 16;
    const f16* Arow = Sh + ((long)(g*PROWS + row0 + lr)) * DD;
    int n0 = np*32 + lr;
    const f16* Brow0 = W1hT + ((long)(g*DD + n0)) * DD;   // [n][k], contiguous in k
    const f16* Brow1 = Brow0 + (long)16 * DD;
    __builtin_prefetch(Arow, 0, 0);
    __builtin_prefetch(Brow0, 0, 0);
    __builtin_prefetch(Brow1, 0, 0);
    v8f c0 = {0.f,0.f,0.f,0.f,0.f,0.f,0.f,0.f};
    v8f c1 = {0.f,0.f,0.f,0.f,0.f,0.f,0.f,0.f};
    #pragma unroll 4
    for (int ks = 0; ks < 16; ++ks) {
        int k0 = ks*32 + 8*half;             // two contiguous runs of 8 per lane
        v16h a  = frag16(Arow, k0);
        v16h b0 = frag16(Brow0, k0);
        v16h b1 = frag16(Brow1, k0);
        c0 = __builtin_amdgcn_wmma_f32_16x16x32_f16(false, a, false, b0, (short)0, c0, false, false);
        c1 = __builtin_amdgcn_wmma_f32_16x16x32_f16(false, a, false, b1, (short)0, c1, false, false);
    }
    float* Pout = P + ((long)(g*PROWS + row0)) * DD + n0;
    #pragma unroll
    for (int v = 0; v < 8; ++v) {
        Pout[(long)(v + 8*half) * DD]      = c0[v];
        Pout[(long)(v + 8*half) * DD + 16] = c1[v];
    }
}

// ---------------- K8: fused pair-MLP: h1 -> h2 (WMMA) -> logits -> sigmoid ----------------
// one block per (b, move); LDS: W2^T f16 (padded) + h1 tile f16 + logits
#define H1STRIDE 520
#define W2STRIDE 520
#define SHM_BYTES (D2*W2STRIDE*2 + 16*H1STRIDE*2 + 64)
__global__ __launch_bounds__(256) void fused_kernel(const float* __restrict__ P,
                                                    const f16* __restrict__ W2hT,
                                                    const float* __restrict__ b1,
                                                    const float* __restrict__ b2,
                                                    const float* __restrict__ Wh,
                                                    const float* __restrict__ bh,
                                                    float* __restrict__ out, int out_size) {
    extern __shared__ char smem[];
    f16*   W2s    = (f16*)smem;                               // [256][W2STRIDE]
    f16*   h1s    = (f16*)(smem + D2*W2STRIDE*2);             // [16][H1STRIDE]
    float* logits = (float*)(smem + D2*W2STRIDE*2 + 16*H1STRIDE*2);  // [16]

    int t  = threadIdx.x;
    int bm = blockIdx.x;            // 0..151
    int bb = bm / (RR-1);
    const float* P1 = P + ((long)bm * RR) * DD;
    const float* P2 = P + ((long)(PROWS + bm * RR)) * DD;

    // stage W2^T (f16) into padded LDS rows: 256 rows x 64 uint4
    {
        const uint4* src = (const uint4*)W2hT;
        for (int i = t; i < 16384; i += 256) {
            int n = i >> 6;
            int cchunk = i & 63;
            *(uint4*)(W2s + (long)n*W2STRIDE + cchunk*8) = src[i];
        }
    }
    __syncthreads();

    int lane = t & 31;
    int wave = t >> 5;
    int half = lane >> 4;
    int lr   = lane & 15;

    for (int mt = 0; mt < 25; ++mt) {
        // ---- build h1 tile (16 rows x 512) in LDS as f16 ----
        {
            int rloc = t >> 4;                 // 0..15
            int rr = mt*16 + rloc;             // 0..399 exactly
            int i = rr / RR, j = rr % RR;
            int k0 = (t & 15) * 32;
            const float4* p1 = (const float4*)(P1 + (long)i*DD + k0);
            const float4* p2 = (const float4*)(P2 + (long)j*DD + k0);
            f16* hrow = h1s + rloc*H1STRIDE + k0;
            #pragma unroll
            for (int q = 0; q < 8; ++q) {
                float4 u = p1[q];
                float4 v = p2[q];
                int kb = k0 + 4*q;
                hrow[4*q+0] = (f16)lrelu01(u.x + v.x + b1[kb+0]);
                hrow[4*q+1] = (f16)lrelu01(u.y + v.y + b1[kb+1]);
                hrow[4*q+2] = (f16)lrelu01(u.z + v.z + b1[kb+2]);
                hrow[4*q+3] = (f16)lrelu01(u.w + v.w + b1[kb+3]);
            }
        }
        if (t < 16) logits[t] = 0.f;
        __syncthreads();

        // ---- h2 = lrelu(h1 @ W2 + b2); dot with Wh, reduce ----
        // wave handles N columns [wave*32, wave*32+32): two tiles, one shared A fragment
        const f16* h1row = h1s + lr*H1STRIDE;
        int n0 = wave*32 + lr;
        const f16* Brow0 = W2s + (long)n0*W2STRIDE;
        const f16* Brow1 = Brow0 + (long)16*W2STRIDE;
        v8f c0 = {0.f,0.f,0.f,0.f,0.f,0.f,0.f,0.f};
        v8f c1 = {0.f,0.f,0.f,0.f,0.f,0.f,0.f,0.f};
        #pragma unroll 4
        for (int ks = 0; ks < 16; ++ks) {
            int k0 = ks*32 + 8*half;
            v16h a  = frag16(h1row, k0);
            v16h b0 = frag16(Brow0, k0);
            v16h b1 = frag16(Brow1, k0);
            c0 = __builtin_amdgcn_wmma_f32_16x16x32_f16(false, a, false, b0, (short)0, c0, false, false);
            c1 = __builtin_amdgcn_wmma_f32_16x16x32_f16(false, a, false, b1, (short)0, c1, false, false);
        }
        #pragma unroll
        for (int sub = 0; sub < 2; ++sub) {
            int n = n0 + sub*16;
            float whn = Wh[n];
            float b2n = b2[n];
            #pragma unroll
            for (int v = 0; v < 8; ++v) {
                float h2 = lrelu01((sub ? c1[v] : c0[v]) + b2n);
                float contrib = h2 * whn;
                contrib += __shfl_xor(contrib, 1, 32);
                contrib += __shfl_xor(contrib, 2, 32);
                contrib += __shfl_xor(contrib, 4, 32);
                contrib += __shfl_xor(contrib, 8, 32);
                if (lr == 0) atomicAdd(&logits[v + 8*half], contrib);
            }
        }
        __syncthreads();

        if (t < 16) {
            int rr = mt*16 + t;
            float logit = logits[t] + bh[0];
            float prob = 1.f / (1.f + __expf(-logit));
            int gidx = bm * (RR*RR) + rr;
            out[gidx] = prob;
            if (NOUT + gidx < out_size) out[NOUT + gidx] = (float)bb;
        }
        __syncthreads();
    }
}

extern "C" void kernel_launch(void* const* d_in, const int* in_sizes, int n_in,
                              void* d_out, int out_size, void* d_ws, size_t ws_size,
                              hipStream_t stream) {
    const float* x1    = (const float*)d_in[0];
    const float* x2    = (const float*)d_in[1];
    const float* Wgat  = (const float*)d_in[2];
    const float* att_s = (const float*)d_in[3];
    const float* att_d = (const float*)d_in[4];
    const float* b_gat = (const float*)d_in[5];
    const float* W1    = (const float*)d_in[6];
    const float* b1    = (const float*)d_in[7];
    const float* W2    = (const float*)d_in[8];
    const float* b2    = (const float*)d_in[9];
    const float* Wh    = (const float*)d_in[10];
    const float* bh    = (const float*)d_in[11];
    const int* ei1     = (const int*)d_in[12];
    const int* ei2     = (const int*)d_in[13];
    const int* rt1     = (const int*)d_in[14];
    const int* rt2     = (const int*)d_in[15];
    float* out = (float*)d_out;

    char* ws = (char*)d_ws;
    float*    h     = (float*)(ws + OFF_H);
    float*    asrc  = (float*)(ws + OFF_ASRC);
    float*    adst  = (float*)(ws + OFF_ADST);
    unsigned* mkey  = (unsigned*)(ws + OFF_MKEY);
    float*    Z     = (float*)(ws + OFF_Z);
    float*    ebuf  = (float*)(ws + OFF_EBUF);
    float*    gout  = (float*)(ws + OFF_GOUT);
    float*    r     = (float*)(ws + OFF_R);
    f16*      sh    = (f16*)(ws + OFF_S);
    float*    P     = (float*)(ws + OFF_P);
    f16*      W1hT  = (f16*)(ws + OFF_W1H);
    f16*      W2hT  = (f16*)(ws + OFF_W2H);

    // zero accumulators (mkey zero == -inf under monotone key)
    hipMemsetAsync(ws + OFF_MKEY, 0, 512000, stream);                 // mkey + Z (contiguous)
    hipMemsetAsync(ws + OFF_GOUT, 0, 16384000 + 655360, stream);      // gout + r (contiguous)

    convert_weights<<<(1024*512 + 255)/256, 256, 0, stream>>>(W1, W2, W1hT, W2hT);
    node_kernel<<<2*NT, 512, 0, stream>>>(x1, x2, Wgat, att_s, att_d, h, asrc, adst);

    int eThreads = 2*EDG*HH;
    edge_a_kernel<<<(eThreads + 255)/256, 256, 0, stream>>>(ei1, ei2, asrc, adst, ebuf, mkey);
    edge_b_kernel<<<(eThreads + 255)/256, 256, 0, stream>>>(ei1, ei2, ebuf, mkey, Z);
    long cThreads = (long)2*EDG*128;
    edge_c_kernel<<<(int)((cThreads + 255)/256), 256, 0, stream>>>(ei1, ei2, ebuf, Z, h, gout);

    emb_pool_kernel<<<(2*NT*DD + 255)/256, 256, 0, stream>>>(gout, b_gat, rt1, rt2, r);
    winsum_kernel<<<(2*BB*DD + 255)/256, 256, 0, stream>>>(r, sh);

    pgemm_kernel<<<760, 256, 0, stream>>>(sh, W1hT, P);

    hipFuncSetAttribute((const void*)fused_kernel,
                        hipFuncAttributeMaxDynamicSharedMemorySize, SHM_BYTES);
    fused_kernel<<<BB*(RR-1), 256, SHM_BYTES, stream>>>(P, W2hT, b1, b2, Wh, bh, out, out_size);
}
